// Weight_Assigner_15710990369327
// MI455X (gfx1250) — matrix operand order repaired and checked by reading, not compiled, not run on hardware
//
#include <hip/hip_runtime.h>
#include <math.h>

#define NCLASS 64
#define TOPK   16
#define NHID   128
#define DEGREE 6

typedef __attribute__((ext_vector_type(16))) _Float16 v16h;
typedef __attribute__((ext_vector_type(8)))  float    v8f;

// Async global -> LDS copy of `ndw` dwords using the gfx1250 async-copy path.
// LDS flat pointers truncate to the raw LDS byte offset (ISA 10.2: low 32 address bits).
__device__ __forceinline__ void async_copy_dw(void* lds_dst, const float* src,
                                              int ndw, int tid) {
    unsigned base = (unsigned)(unsigned long long)lds_dst;
    for (int i = tid; i < ndw; i += 256) {
        unsigned     laddr = base + (unsigned)i * 4u;
        const float* gaddr = src + i;
        asm volatile("global_load_async_to_lds_b32 %0, %1, off"
                     :: "v"(laddr), "v"(gaddr) : "memory");
    }
}

// ---------------------------------------------------------------------------
// Gate kernel: per-row softmax -> sorted top-16 -> [16->128] WMMA matmul with
// leaky-relu -> [128->6] -> softmax -> per-row hop weights.
// Block = 256 threads = 8 waves; each wave owns 16 rows (block: 128 rows).
// ---------------------------------------------------------------------------
__global__ __launch_bounds__(256) void gate_kernel(
    const float* __restrict__ x, const float* __restrict__ W1,
    const float* __restrict__ b1, const float* __restrict__ W2,
    const float* __restrict__ b2, float* __restrict__ gate, int n)
{
    __shared__ float    s_rawW1[NHID * TOPK];   // async-staged raw W1 (8 KB)
    __shared__ float    s_rawW2[DEGREE * NHID]; // async-staged raw W2 (3 KB)
    __shared__ _Float16 s_b1[32][NHID];         // W1^T, K padded 16->32 (8 KB)
    __shared__ float    s_w2[DEGREE][NHID];     // 3 KB
    __shared__ float    s_bias1[NHID];
    __shared__ float    s_bias2[8];
    __shared__ _Float16 s_a1[8][16][32];        // per-wave A tile (top-k probs) 8 KB
    __shared__ _Float16 s_h[8][16][NHID];       // per-wave hidden acts 32 KB

    const int tid = threadIdx.x;

    // -- async-stage raw weights into LDS (ASYNCcnt-tracked, no VGPR round trip)
    async_copy_dw(s_rawW1, W1, NHID * TOPK, tid);
    async_copy_dw(s_rawW2, W2, DEGREE * NHID, tid);
    asm volatile("s_wait_asynccnt 0x0" ::: "memory");
    __syncthreads();

    // -- convert to WMMA-friendly layouts
    for (int i = tid; i < 32 * NHID; i += 256) {
        int k = i / NHID, nn = i % NHID;
        s_b1[k][nn] = (k < TOPK) ? (_Float16)s_rawW1[nn * TOPK + k] : (_Float16)0.0f;
    }
    for (int i = tid; i < DEGREE * NHID; i += 256) s_w2[i / NHID][i % NHID] = s_rawW2[i];
    for (int i = tid; i < NHID; i += 256) s_bias1[i] = b1[i];
    if (tid < DEGREE) s_bias2[tid] = b2[tid];
    __syncthreads();

    const int wave = tid >> 5;
    const int lane = tid & 31;
    const int rowbase = blockIdx.x * 128 + wave * 16;

    // ---- phase 1: softmax over 64 classes + sorted top-16 selection ----
    for (int t = 0; t < 16; ++t) {
        int row = rowbase + t;
        int rr  = row < n ? row : (n - 1);
        float v0 = x[(long long)rr * NCLASS + lane];
        float v1 = x[(long long)rr * NCLASS + 32 + lane];
        float m = fmaxf(v0, v1);
        #pragma unroll
        for (int off = 16; off > 0; off >>= 1) m = fmaxf(m, __shfl_xor(m, off, 32));
        float e0 = __expf(v0 - m), e1 = __expf(v1 - m);
        float s = e0 + e1;
        #pragma unroll
        for (int off = 16; off > 0; off >>= 1) s += __shfl_xor(s, off, 32);
        float p0 = e0 / s, p1 = e1 / s;
        if (lane >= 16) s_a1[wave][t][lane] = (_Float16)0.0f;  // zero-pad K 16..31
        for (int j = 0; j < TOPK; ++j) {   // selection sort: descending order
            float bv; int bi;
            if (p0 >= p1) { bv = p0; bi = lane; } else { bv = p1; bi = lane + 32; }
            #pragma unroll
            for (int off = 16; off > 0; off >>= 1) {
                float ov = __shfl_xor(bv, off, 32);
                int   oi = __shfl_xor(bi, off, 32);
                if (ov > bv || (ov == bv && oi < bi)) { bv = ov; bi = oi; }
            }
            if (lane == (bi & 31)) { if (bi < 32) p0 = -1.0f; else p1 = -1.0f; }
            if (lane == j) s_a1[wave][t][j] = (_Float16)bv;
        }
    }

    // ---- phase 2: h = leaky_relu(tk @ W1^T + b1) via v_wmma_f32_16x16x32_f16 ----
    const int g  = lane >> 4;   // half-wave selector
    const int ml = lane & 15;   // M (for A) / N (for B) position
    v16h a;
    #pragma unroll
    for (int v = 0; v < 8; ++v) {
        // 16-bit A 16x32 layout: lanes<16: K=0..7 & 16..23 ; lanes>=16: K=8..15 & 24..31
        int k = g * 8 + (v < 4 ? 2 * v : 16 + 2 * (v - 4));
        a[2 * v]     = s_a1[wave][ml][k];
        a[2 * v + 1] = s_a1[wave][ml][k + 1];
    }
    #pragma unroll
    for (int nt = 0; nt < 8; ++nt) {
        v16h b;
        #pragma unroll
        for (int v = 0; v < 8; ++v) {
            // B 32x16 layout: lanes<16: K=0..7 & 16..23 ; lanes>=16: K=8..15 & 24..31
            int k = g * 8 + (v < 4 ? 2 * v : 16 + 2 * (v - 4));
            b[2 * v]     = s_b1[k][nt * 16 + ml];
            b[2 * v + 1] = s_b1[k + 1][nt * 16 + ml];
        }
        v8f c = {};
        c = __builtin_amdgcn_wmma_f32_16x16x32_f16(false, a, false, b,
                                                   (short)0, c, false, false);
        #pragma unroll
        for (int j = 0; j < 8; ++j) {
            int M  = j + g * 8;              // C/D layout: vgpr j -> M=j / M=j+8
            int nn = nt * 16 + ml;
            float hv = c[j] + s_bias1[nn];
            hv = hv > 0.0f ? hv : 0.1f * hv; // leaky_relu(0.1)
            s_h[wave][M][nn] = (_Float16)hv;
        }
    }

    // ---- phase 3: weight = softmax(h @ W2^T + b2), 2 lanes x 3 outputs per row ----
    const int r  = lane & 15;
    const int hh = lane >> 4;
    float acc[3];
    #pragma unroll
    for (int d = 0; d < 3; ++d) {
        int dd = hh * 3 + d;
        float sacc = s_bias2[dd];
        for (int cidx = 0; cidx < NHID; ++cidx)
            sacc += (float)s_h[wave][r][cidx] * s_w2[dd][cidx];
        acc[d] = sacc;
    }
    float mx = fmaxf(acc[0], fmaxf(acc[1], acc[2]));
    mx = fmaxf(mx, __shfl_xor(mx, 16, 32));
    float es = 0.0f;
    #pragma unroll
    for (int d = 0; d < 3; ++d) { acc[d] = __expf(acc[d] - mx); es += acc[d]; }
    es += __shfl_xor(es, 16, 32);
    int grow = rowbase + r;
    if (grow < n) {
        #pragma unroll
        for (int d = 0; d < 3; ++d)
            gate[(long long)grow * DEGREE + hh * 3 + d] = acc[d] / es;
    }
}

// ---------------------------------------------------------------------------
// SpMM: one wave per edge. Lane i moves float2 (cols 2i,2i+1) of src row,
// scaled native-FP32-atomic-add into dst row. 256B rows, fully coalesced.
// ---------------------------------------------------------------------------
__global__ __launch_bounds__(256) void spmm_kernel(
    const int* __restrict__ er, const int* __restrict__ ec,
    const float* __restrict__ ev, const float* __restrict__ cur,
    float* __restrict__ nxt, long long ne)
{
    long long e = (long long)blockIdx.x * 8 + (threadIdx.x >> 5);
    if (e >= ne) return;                       // wave-uniform exit
    const int lane = threadIdx.x & 31;
    const int rr = er[e];
    const int cc = ec[e];
    const float v = ev[e];
    const float2 d = ((const float2*)(cur + (long long)cc * NCLASS))[lane];
    float* dst = nxt + (long long)rr * NCLASS + lane * 2;
    float ax = v * d.x, ay = v * d.y;
    // guaranteed single-instruction L2 FP32 RMW (no CAS expansion)
    asm volatile("global_atomic_add_f32 %0, %1, off\n\t"
                 "global_atomic_add_f32 %0, %2, off offset:4"
                 :: "v"(dst), "v"(ax), "v"(ay) : "memory");
}

__global__ void init_kernel(float* __restrict__ out, const float* __restrict__ gate,
                            const float* __restrict__ x, long long total)
{
    long long i = (long long)blockIdx.x * blockDim.x + threadIdx.x;
    if (i >= total) return;
    out[i] = gate[(i >> 6) * DEGREE] * x[i];
}

__global__ void axpy_kernel(float* __restrict__ out, const float* __restrict__ gate,
                            const float* __restrict__ cur, int hop, long long total)
{
    long long i = (long long)blockIdx.x * blockDim.x + threadIdx.x;
    if (i >= total) return;
    out[i] += gate[(i >> 6) * DEGREE + hop] * cur[i];
}

// log_softmax over 64 classes; one wave per row, 2 values per lane.
__global__ __launch_bounds__(256) void lsm_kernel(float* __restrict__ out, int n)
{
    int row = blockIdx.x * 8 + (threadIdx.x >> 5);
    if (row >= n) return;                      // wave-uniform exit
    int lane = threadIdx.x & 31;
    long long base = (long long)row * NCLASS;
    float v0 = out[base + lane], v1 = out[base + 32 + lane];
    float m = fmaxf(v0, v1);
    #pragma unroll
    for (int off = 16; off > 0; off >>= 1) m = fmaxf(m, __shfl_xor(m, off, 32));
    float s = __expf(v0 - m) + __expf(v1 - m);
    #pragma unroll
    for (int off = 16; off > 0; off >>= 1) s += __shfl_xor(s, off, 32);
    float l = m + __logf(s);
    out[base + lane]      = v0 - l;
    out[base + 32 + lane] = v1 - l;
}

extern "C" void kernel_launch(void* const* d_in, const int* in_sizes, int n_in,
                              void* d_out, int out_size, void* d_ws, size_t ws_size,
                              hipStream_t stream)
{
    const float* x  = (const float*)d_in[0];
    const int*   er = (const int*)d_in[1];
    const int*   ec = (const int*)d_in[2];
    const float* ev = (const float*)d_in[3];
    const float* W1 = (const float*)d_in[4];
    const float* b1 = (const float*)d_in[5];
    const float* W2 = (const float*)d_in[6];
    const float* b2 = (const float*)d_in[7];
    const int n = in_sizes[0] / NCLASS;
    const long long ne = in_sizes[1];
    float* out = (float*)d_out;

    // workspace carve-out (256B aligned)
    char* ws = (char*)d_ws;
    size_t off = 0;
    auto carve = [&](size_t bytes) -> float* {
        float* p = (float*)(ws + off);
        off = (off + bytes + 255) & ~(size_t)255;
        return p;
    };
    float* gate = carve((size_t)n * DEGREE * sizeof(float));
    float* bufA = carve((size_t)n * NCLASS * sizeof(float));
    float* bufB = carve((size_t)n * NCLASS * sizeof(float));

    gate_kernel<<<(n + 127) / 128, 256, 0, stream>>>(x, W1, b1, W2, b2, gate, n);

    const long long total = (long long)n * NCLASS;
    const unsigned gelem = (unsigned)((total + 255) / 256);
    init_kernel<<<gelem, 256, 0, stream>>>(out, gate, x, total);

    const float* cur = x;
    float* bufs[2] = {bufA, bufB};
    for (int hop = 1; hop < DEGREE; ++hop) {
        float* nxt = bufs[(hop - 1) & 1];
        hipMemsetAsync(nxt, 0, (size_t)n * NCLASS * sizeof(float), stream);
        spmm_kernel<<<(unsigned)((ne + 7) / 8), 256, 0, stream>>>(er, ec, ev, cur, nxt, ne);
        axpy_kernel<<<gelem, 256, 0, stream>>>(out, gate, nxt, hop, total);
        cur = nxt;
    }
    lsm_kernel<<<(unsigned)((n + 7) / 8), 256, 0, stream>>>(out, n);
}
